// HopfieldASAAttention_14817637171724
// MI455X (gfx1250) — matrix operand order repaired and checked by reading, not compiled, not run on hardware
//
#include <hip/hip_runtime.h>
#include <hip/hip_bf16.h>
#include <math.h>

// Problem constants (match reference)
#define BB 2
#define NN 4096
#define DD 1024
#define PP 64
#define RR 32
#define FF 64              // 2R feature width of phi
#define RFF_SCALE 0.25f    // sqrt(2/32)
#define DCHUNK 128         // output-dim chunk per wave in pass 2

// GEMM blocking
#define GEMM_BM 128        // block rows   (8 waves x 16)
#define GEMM_BN 64         // block cols   (4 wmma tiles per wave)
#define GEMM_KC 128        // k-chunk staged in LDS (64x128 bf16 = 16KB)

typedef __bf16 bf16;
typedef __attribute__((ext_vector_type(16))) __bf16 v16bf;
typedef __attribute__((ext_vector_type(8)))  __bf16 v8bf;
typedef __attribute__((ext_vector_type(8)))  float  v8f;
typedef __attribute__((ext_vector_type(4)))  unsigned int v4u;
typedef __attribute__((ext_vector_type(8)))  int v8i;
typedef __attribute__((ext_vector_type(4)))  int v4i;

#if __has_builtin(__builtin_amdgcn_tensor_load_to_lds)
#define HAVE_TDM 1
#else
#define HAVE_TDM 0
#endif

static __device__ inline float softplus_dev(float x){ return log1pf(__expf(x)); }

// ---------------------------------------------------------------------------
// WMMA fragment loaders (CDNA5 ISA 7.12.2, wave32)
// A 16x32 bf16 per lane: elems0..7 = K[k0+8h .. +7], elems8..15 = K[k0+16+8h .. +7]
// ---------------------------------------------------------------------------
static __device__ inline v16bf frag_a(const bf16* src, int ld, int lane){
  const int h = lane >> 4;
  const bf16* p = src + (size_t)(lane & 15) * ld + h * 8;
  v8bf lo = *(const v8bf*)(p);
  v8bf hi = *(const v8bf*)(p + 16);
  v16bf a;
#pragma unroll
  for (int i = 0; i < 8; ++i){ a[i] = lo[i]; a[i+8] = hi[i]; }
  return a;
}

// B 32x16 bf16 with B[k][n] = W[n0+n][k0+k] for row-major W (row stride ld):
// lane column = lane&15, elems0..7 = K[k0+8h .. +7], elems8..15 = K[k0+16+8h .. +7]
static __device__ inline v16bf frag_b(const bf16* src, int ld, int lane){
  const int h = lane >> 4;
  const bf16* p = src + (size_t)(lane & 15) * ld + h * 8;
  v8bf lo = *(const v8bf*)(p);
  v8bf hi = *(const v8bf*)(p + 16);
  v16bf b;
#pragma unroll
  for (int i = 0; i < 8; ++i){ b[i] = lo[i]; b[i+8] = hi[i]; }
  return b;
}

static __device__ inline float rmax16(float v){
#pragma unroll
  for (int m = 8; m; m >>= 1) v = fmaxf(v, __shfl_xor(v, m, 16));
  return v;
}
static __device__ inline float rsum16(float v){
#pragma unroll
  for (int m = 8; m; m >>= 1) v += __shfl_xor(v, m, 16);
  return v;
}

#define WMMA_BF16(A, Bv, C) \
  __builtin_amdgcn_wmma_f32_16x16x32_bf16(false, (A), false, (Bv), (short)0, (C), false, false)

// ---------------------------------------------------------------------------
// TDM: stage a 2D tile (tile_k elems x tile_n rows, bf16) from row-major global
// (row stride = row_stride elems) into LDS at lds_off. D# per ISA 8.3-8.6.
// Best-effort descriptor packing; guarded fallback below.
// ---------------------------------------------------------------------------
#if HAVE_TDM
static __device__ inline void tdm_load_tile_bf16(unsigned lds_off, const bf16* gsrc,
                                                 unsigned tile_k, unsigned tile_n,
                                                 unsigned tensor_k, unsigned tensor_n,
                                                 unsigned row_stride){
  unsigned long long ga = (unsigned long long)(size_t)gsrc;
  v4u g0;
  g0[0] = 1u;                                            // count=1, user D#
  g0[1] = lds_off;                                       // lds_addr (bytes)
  g0[2] = (unsigned)(ga & 0xffffffffu);                  // global_addr[31:0]
  g0[3] = (unsigned)((ga >> 32) & 0x01ffffffu) | (2u << 30);  // addr[56:32] | type=2
  v8i g1;
  g1[0] = (int)(1u << 16);                               // data_size=1 (2 bytes)
  g1[1] = (int)((tensor_k & 0xffffu) << 16);             // tensor_dim0[15:0]
  g1[2] = (int)(((tensor_k >> 16) & 0xffffu) | ((tensor_n & 0xffffu) << 16)); // dim0 hi | dim1 lo
  g1[3] = (int)(((tensor_n >> 16) & 0xffffu) | ((tile_k & 0xffffu) << 16));   // dim1 hi | tile_dim0
  g1[4] = (int)(tile_n & 0xffffu);                       // tile_dim1 (tile_dim2=0)
  g1[5] = (int)row_stride;                               // tensor_dim0_stride[31:0]
  g1[6] = 0;                                             // stride hi | dim1_stride lo
  g1[7] = 0;
  v4i gz = {0,0,0,0};
#if defined(__clang_major__) && (__clang_major__ >= 23)
  v8i gz8 = {0,0,0,0,0,0,0,0};
  __builtin_amdgcn_tensor_load_to_lds(g0, g1, gz, gz, gz8, 0);
#else
  __builtin_amdgcn_tensor_load_to_lds(g0, g1, gz, gz, 0);
#endif
}
#endif

// ---------------------------------------------------------------------------
// K0: per-token features. One 128-thread block per token.
// ---------------------------------------------------------------------------
__global__ void feat_kernel(const float* __restrict__ x,
                            const float* __restrict__ pos_w, const float* __restrict__ pos_b,
                            const float* __restrict__ chg_w, const float* __restrict__ chg_b,
                            const float* __restrict__ omega, const float* __restrict__ rff_b,
                            const float* __restrict__ cscale, const float* __restrict__ dscale,
                            bf16* __restrict__ xb, bf16* __restrict__ phi, float* __restrict__ ch)
{
  __shared__ float xs[DD];
  __shared__ float pos[PP];
  const int tok = blockIdx.x;
  const int tid = threadIdx.x;
  const float* xr = x + (size_t)tok * DD;
  for (int i = tid; i < DD; i += blockDim.x){
    float f = xr[i];
    xs[i] = f;
    xb[(size_t)tok * DD + i] = (bf16)f;
  }
  __syncthreads();
  if (tid < PP){
    const float* w = pos_w + (size_t)tid * DD;
    float acc = pos_b[tid];
#pragma unroll 4
    for (int d = 0; d < DD; ++d) acc = fmaf(xs[d], w[d], acc);
    pos[tid] = acc;
  } else if (tid == PP){
    float acc = chg_b[0];
#pragma unroll 4
    for (int d = 0; d < DD; ++d) acc = fmaf(xs[d], chg_w[d], acc);
    ch[tok] = tanhf(acc) * sqrtf(softplus_dev(cscale[0]));
  }
  __syncthreads();
  if (tid < RR){
    const float* om = omega + (size_t)tid * PP;
    float pr = rff_b[tid];
#pragma unroll
    for (int p = 0; p < PP; ++p) pr = fmaf(pos[p], om[p], pr);
    const float s = RFF_SCALE * sqrtf(softplus_dev(dscale[0]));
    phi[(size_t)tok * FF + tid]      = (bf16)(__cosf(pr) * s);
    phi[(size_t)tok * FF + RR + tid] = (bf16)(__sinf(pr) * s);
  }
}

__global__ void cvt_bf16_kernel(const float* __restrict__ in, bf16* __restrict__ out, int n){
  int i = blockIdx.x * blockDim.x + threadIdx.x;
  if (i < n) out[i] = (bf16)in[i];
}

// ---------------------------------------------------------------------------
// K1: C = A[M,K] * W[Nout,K]^T + bias. Block tile 128x64, 8 waves, each wave
// computes a 16x64 strip (4 accumulators; A-fragment reused 4x). The shared
// 64x128 W k-chunk is staged in LDS (TDM if available, else vector copy).
// MODE 0: fp32 row-major output. MODE 1: bf16 transposed [Nout][M] output.
// ---------------------------------------------------------------------------
template<int MODE>
__global__ void gemm_kernel(const bf16* __restrict__ A, const bf16* __restrict__ W,
                            const float* __restrict__ bias, float* __restrict__ Cf,
                            bf16* __restrict__ Ct, int M, int Nout, int K)
{
  __shared__ bf16 Wl[GEMM_BN * GEMM_KC];    // 16 KB
  const int lane = threadIdx.x & 31;
  const int wv   = threadIdx.x >> 5;        // 0..7
  const int tilesN = Nout / GEMM_BN;
  const int m0 = (blockIdx.x / tilesN) * GEMM_BM + wv * 16;
  const int n0 = (blockIdx.x % tilesN) * GEMM_BN;
  const bf16* Ab = A + (size_t)blockIdx.y * M * K + (size_t)m0 * K;

  v8f acc[4];
#pragma unroll
  for (int t = 0; t < 4; ++t) acc[t] = (v8f){0.f,0.f,0.f,0.f,0.f,0.f,0.f,0.f};

  for (int kc = 0; kc < K; kc += GEMM_KC){
    // ---- stage W[n0 .. n0+63][kc .. kc+127] into LDS ----
#if HAVE_TDM
    if (wv == 0){
      tdm_load_tile_bf16((unsigned)(size_t)&Wl[0],
                         W + (size_t)n0 * K + kc,
                         GEMM_KC, GEMM_BN, (unsigned)K, (unsigned)Nout, (unsigned)K);
      __builtin_amdgcn_s_wait_tensorcnt(0);
    }
#else
    for (int idx = threadIdx.x; idx < (GEMM_BN * GEMM_KC) / 8; idx += blockDim.x){
      const int r = idx / (GEMM_KC / 8);
      const int c = (idx % (GEMM_KC / 8)) * 8;
      *(v8bf*)&Wl[r * GEMM_KC + c] = *(const v8bf*)&W[(size_t)(n0 + r) * K + kc + c];
    }
#endif
    __syncthreads();

#pragma unroll
    for (int ks = 0; ks < GEMM_KC; ks += 32){
      __builtin_prefetch(Ab + kc + ks + 512, 0, 0);
      v16bf a = frag_a(Ab + kc + ks, K, lane);
#pragma unroll
      for (int t = 0; t < 4; ++t){
        v16bf bv = frag_b(&Wl[(t * 16) * GEMM_KC + ks], GEMM_KC, lane);
        acc[t] = WMMA_BF16(a, bv, acc[t]);
      }
    }
    __syncthreads();
  }

  const int rbase = m0 + ((lane >> 4) << 3);
#pragma unroll
  for (int t = 0; t < 4; ++t){
    const int col = n0 + t * 16 + (lane & 15);
    const float bc = bias[col];
    if (MODE == 0){
#pragma unroll
      for (int i = 0; i < 8; ++i)
        Cf[(size_t)(rbase + i) * Nout + col] = acc[t][i] + bc;
    } else {
      bf16* Cb = Ct + (size_t)blockIdx.y * Nout * M;
#pragma unroll
      for (int i = 0; i < 8; ++i)
        Cb[(size_t)col * M + (rbase + i)] = (bf16)(acc[t][i] + bc);
    }
  }
}

// ---------------------------------------------------------------------------
// K2: flash pass 1 — online softmax stats over causal key tiles.
// ---------------------------------------------------------------------------
__global__ void stats_kernel(const bf16* __restrict__ phi, const float* __restrict__ ch,
                             float* __restrict__ mbuf, float* __restrict__ lbuf)
{
  const int lane = threadIdx.x & 31;
  const int wid  = blockIdx.x * (blockDim.x >> 5) + (threadIdx.x >> 5);
  const int b  = wid / (NN >> 4);
  const int m0 = (wid % (NN >> 4)) << 4;
  const bf16*  phb = phi + (size_t)b * NN * FF;
  const float* chb = ch  + (size_t)b * NN;
  const int h = lane >> 4, col = lane & 15;

  const v16bf a0 = frag_a(phb + (size_t)m0 * FF +  0, FF, lane);
  const v16bf a1 = frag_a(phb + (size_t)m0 * FF + 32, FF, lane);

  float chRow[8], Mr[8], Lr[8];
#pragma unroll
  for (int i = 0; i < 8; ++i){
    chRow[i] = chb[m0 + i + 8*h];
    Mr[i] = -INFINITY; Lr[i] = 0.0f;
  }

  for (int j0 = 0; j0 <= m0; j0 += 16){
    v16bf b0 = frag_b(phb + (size_t)j0 * FF +  0, FF, lane);
    v16bf b1 = frag_b(phb + (size_t)j0 * FF + 32, FF, lane);
    v8f s = {0.f,0.f,0.f,0.f,0.f,0.f,0.f,0.f};
    s = WMMA_BF16(a0, b0, s);
    s = WMMA_BF16(a1, b1, s);
    const float chc  = chb[j0 + col];
    const int   colg = j0 + col;
#pragma unroll
    for (int i = 0; i < 8; ++i){
      const int rowg = m0 + i + 8*h;
      float v = s[i] - chRow[i] * chc;
      if (colg > rowg) v = -INFINITY;
      const float tmax = rmax16(v);
      const float nM   = fmaxf(Mr[i], tmax);
      const float e    = __expf(v - nM);
      const float ts   = rsum16(e);
      Lr[i] = Lr[i] * __expf(Mr[i] - nM) + ts;
      Mr[i] = nM;
    }
  }
  if (col == 0){
#pragma unroll
    for (int i = 0; i < 8; ++i){
      const int rowg = m0 + i + 8*h;
      mbuf[(size_t)b * NN + rowg] = Mr[i];
      lbuf[(size_t)b * NN + rowg] = Lr[i];
    }
  }
}

// ---------------------------------------------------------------------------
// K3: flash pass 2 — recompute S tiles, P=exp(S-m)/l (bf16 via LDS), then
// O[16 x DCHUNK] += P(16x32) x Vt(32x16 per subtile) via WMMA.
// Consecutive waves in a block share the same D-chunk -> Vt L2 locality.
// ---------------------------------------------------------------------------
__global__ void pass2_kernel(const bf16* __restrict__ phi, const float* __restrict__ ch,
                             const bf16* __restrict__ Vt,  const float* __restrict__ mbuf,
                             const float* __restrict__ lbuf, bf16* __restrict__ Obf)
{
  __shared__ bf16 Pt[8][16 * 32];
  const int lane  = threadIdx.x & 31;
  const int wslot = threadIdx.x >> 5;
  const int wid   = blockIdx.x * (blockDim.x >> 5) + wslot;
  const int tilesD = DD / DCHUNK;      // 8
  const int tilesM = NN >> 4;          // 256
  const int b   = wid / (tilesM * tilesD);
  const int rem = wid % (tilesM * tilesD);
  const int m0  = (rem % tilesM) << 4;       // row tile fastest -> block shares d0
  const int d0  = (rem / tilesM) * DCHUNK;
  const bf16*  phb = phi + (size_t)b * NN * FF;
  const float* chb = ch  + (size_t)b * NN;
  const bf16*  Vtb = Vt  + (size_t)b * DD * NN;
  const int h = lane >> 4, col = lane & 15;

  const v16bf a0 = frag_a(phb + (size_t)m0 * FF +  0, FF, lane);
  const v16bf a1 = frag_a(phb + (size_t)m0 * FF + 32, FF, lane);

  float chRow[8], Mr[8], rLr[8];
#pragma unroll
  for (int i = 0; i < 8; ++i){
    const int rowg = m0 + i + 8*h;
    chRow[i] = chb[rowg];
    Mr[i]    = mbuf[(size_t)b * NN + rowg];
    rLr[i]   = 1.0f / lbuf[(size_t)b * NN + rowg];
  }

  v8f O[8];
#pragma unroll
  for (int t = 0; t < 8; ++t) O[t] = (v8f){0.f,0.f,0.f,0.f,0.f,0.f,0.f,0.f};

  bf16* Pw = &Pt[wslot][0];
  const int jend = m0 + 16;
  for (int j0 = 0; j0 < jend; j0 += 32){
#pragma unroll
    for (int sub = 0; sub < 2; ++sub){
      const int k0 = j0 + 16 * sub;
      v16bf b0 = frag_b(phb + (size_t)k0 * FF +  0, FF, lane);
      v16bf b1 = frag_b(phb + (size_t)k0 * FF + 32, FF, lane);
      v8f s = {0.f,0.f,0.f,0.f,0.f,0.f,0.f,0.f};
      s = WMMA_BF16(a0, b0, s);
      s = WMMA_BF16(a1, b1, s);
      const float chc  = chb[k0 + col];
      const int   colg = k0 + col;
#pragma unroll
      for (int i = 0; i < 8; ++i){
        const int rowg = m0 + i + 8*h;
        float p = 0.0f;
        if (colg <= rowg){
          const float v = s[i] - chRow[i] * chc;
          p = __expf(v - Mr[i]) * rLr[i];
        }
        Pw[(i + 8*h) * 32 + sub * 16 + col] = (bf16)p;
      }
    }
    // cross-lane LDS reuse within the wave: fence DS stores before fragment load
    asm volatile("s_wait_dscnt 0" ::: "memory");
    __builtin_amdgcn_wave_barrier();

    const v16bf aP = frag_a(Pw, 32, lane);
#pragma unroll
    for (int t = 0; t < 8; ++t){
      v16bf bv = frag_b(Vtb + (size_t)(d0 + 16 * t) * NN + j0, NN, lane);
      O[t] = WMMA_BF16(aP, bv, O[t]);
    }
    __builtin_amdgcn_wave_barrier();
  }

#pragma unroll
  for (int t = 0; t < 8; ++t){
    const int dcol = d0 + 16 * t + col;
#pragma unroll
    for (int i = 0; i < 8; ++i){
      const int rowg = m0 + i + 8*h;
      Obf[((size_t)b * NN + rowg) * DD + dcol] = (bf16)O[t][i];
    }
  }
}

// ---------------------------------------------------------------------------
// Launcher
// ---------------------------------------------------------------------------
extern "C" void kernel_launch(void* const* d_in, const int* in_sizes, int n_in,
                              void* d_out, int out_size, void* d_ws, size_t ws_size,
                              hipStream_t stream)
{
  const float* x      = (const float*)d_in[0];
  const float* pos_w  = (const float*)d_in[1];
  const float* pos_b  = (const float*)d_in[2];
  const float* chg_w  = (const float*)d_in[3];
  const float* chg_b  = (const float*)d_in[4];
  const float* omega  = (const float*)d_in[5];
  const float* rff_b  = (const float*)d_in[6];
  const float* v_w    = (const float*)d_in[7];
  const float* v_b    = (const float*)d_in[8];
  const float* out_w  = (const float*)d_in[9];
  const float* out_b  = (const float*)d_in[10];
  const float* cscale = (const float*)d_in[11];
  const float* dscale = (const float*)d_in[12];
  float* out = (float*)d_out;

  char* ws = (char*)d_ws;
  size_t off = 0;
  auto carve = [&](size_t bytes) -> char* {
    char* p = ws + off;
    off += (bytes + 255) & ~(size_t)255;
    return p;
  };
  bf16*  xb   = (bf16*) carve((size_t)BB * NN * DD * 2);
  bf16*  phi  = (bf16*) carve((size_t)BB * NN * FF * 2);
  float* ch   = (float*)carve((size_t)BB * NN * 4);
  bf16*  vwb  = (bf16*) carve((size_t)DD * DD * 2);
  bf16*  owb  = (bf16*) carve((size_t)DD * DD * 2);
  bf16*  Vt   = (bf16*) carve((size_t)BB * DD * NN * 2);
  float* mbuf = (float*)carve((size_t)BB * NN * 4);
  float* lbuf = (float*)carve((size_t)BB * NN * 4);
  bf16*  Obf  = (bf16*) carve((size_t)BB * NN * DD * 2);
  (void)ws_size; (void)in_sizes; (void)n_in; (void)out_size;

  feat_kernel<<<BB * NN, 128, 0, stream>>>(x, pos_w, pos_b, chg_w, chg_b, omega,
                                           rff_b, cscale, dscale, xb, phi, ch);

  cvt_bf16_kernel<<<(DD * DD + 255) / 256, 256, 0, stream>>>(v_w,  vwb, DD * DD);
  cvt_bf16_kernel<<<(DD * DD + 255) / 256, 256, 0, stream>>>(out_w, owb, DD * DD);

  // K1a: Vt[b] = (xb[b] @ v_w^T + v_b)^T  (bf16 transposed, per batch)
  {
    dim3 grid((NN / GEMM_BM) * (DD / GEMM_BN), BB);  // 32*16 = 512 blocks x 2
    gemm_kernel<1><<<grid, 256, 0, stream>>>(xb, vwb, v_b, nullptr, Vt, NN, DD, DD);
  }

  // K2: softmax stats
  stats_kernel<<<(BB * (NN >> 4)) / 8, 256, 0, stream>>>(phi, ch, mbuf, lbuf);

  // K3: O = softmax(S) @ V
  pass2_kernel<<<(BB * (NN >> 4) * (DD / DCHUNK)) / 8, 256, 0, stream>>>(
      phi, ch, Vt, mbuf, lbuf, Obf);

  // K1b: out = Obf @ out_w^T + out_b  (fp32, batch folded into M)
  {
    dim3 grid(((BB * NN) / GEMM_BM) * (DD / GEMM_BN), 1);  // 64*16 = 1024 blocks
    gemm_kernel<0><<<grid, 256, 0, stream>>>(Obf, owb, out_b, out, nullptr, BB * NN, DD, DD);
  }
}
